// SpanBert_70763881169555
// MI455X (gfx1250) — compile-verified, matrix operand order checked by reading, not run on hardware
//
#include <hip/hip_runtime.h>
#include <hip/hip_bf16.h>
#include <math.h>

// ---------------- problem constants ----------------
#define H_DIM   768
#define SEQ_S   4096
#define NSPAN   8192
#define NBATCH  8
#define TOPK    50
#define PHID    150
#define PHPAD   192     // PAIR_HID padded to multiple of 64 for WMMA N-tiles
#define MAXW    10
#define BT_PITCH 40     // LDS pitch (elems) for transposed B tiles: 80B rows, 16B-aligned runs
#define KTILES  (H_DIM / 32)

typedef __attribute__((ext_vector_type(16))) __bf16 v16bf;
typedef __attribute__((ext_vector_type(8)))  __bf16 v8bf;
typedef __attribute__((ext_vector_type(8)))  float  v8f;
typedef __attribute__((ext_vector_type(4)))  int    v4i;

// -------- CDNA5 async global->LDS copy (ASYNCcnt path) --------
#if __has_builtin(__builtin_amdgcn_global_load_async_to_lds_b128)
#define HAVE_ASYNC_LDS 1
__device__ __forceinline__ void async_copy16(const void* g, void* l) {
  __builtin_amdgcn_global_load_async_to_lds_b128(
      (__attribute__((address_space(1))) v4i*)g,
      (__attribute__((address_space(3))) v4i*)l, 0, 0);
}
__device__ __forceinline__ void async_wait0() {
#if __has_builtin(__builtin_amdgcn_s_wait_asynccnt)
  __builtin_amdgcn_s_wait_asynccnt(0);
#else
  asm volatile("s_wait_asynccnt 0" ::: "memory");
#endif
}
#else
#define HAVE_ASYNC_LDS 0
__device__ __forceinline__ void async_copy16(const void* g, void* l) {
  *(v8bf*)l = *(const v8bf*)g;
}
__device__ __forceinline__ void async_wait0() {}
#endif

// ---------------- bf16 helpers (RNE) ----------------
__device__ __forceinline__ __bf16 f2bf(float x) {
  union { float f; unsigned u; } a; a.f = x;
  unsigned r = a.u + 0x7FFFu + ((a.u >> 16) & 1u);
  union { unsigned short s; __bf16 b; } c; c.s = (unsigned short)(r >> 16);
  return c.b;
}
__device__ __forceinline__ float bf2f(__bf16 b) {
  union { __bf16 b; unsigned short s; } c; c.b = b;
  union { unsigned u; float f; } a; a.u = ((unsigned)c.s) << 16;
  return a.f;
}
__device__ __forceinline__ v8bf pack8(float4 a, float4 b) {
  v8bf r;
  r[0] = f2bf(a.x); r[1] = f2bf(a.y); r[2] = f2bf(a.z); r[3] = f2bf(a.w);
  r[4] = f2bf(b.x); r[5] = f2bf(b.y); r[6] = f2bf(b.z); r[7] = f2bf(b.w);
  return r;
}

// ---------------- WMMA fragment loaders ----------------
// Per ISA 7.12.2 (16-bit 16x32 operand): lane holds two contiguous 8-elem runs,
// K in [kh,kh+8) and [kh+16,kh+24), kh = (lane&16)?8:0 -> 2x ds_load_b128.
__device__ __forceinline__ v16bf load_frag_rows(const __bf16* rowbase, int kh) {
  v8bf lo = *(const v8bf*)(rowbase + kh);
  v8bf hi = *(const v8bf*)(rowbase + kh + 16);
  return __builtin_shufflevector(lo, hi, 0,1,2,3,4,5,6,7,8,9,10,11,12,13,14,15);
}
// A tile: row-major [rows][32]; lane row = tileRow + (lane&15)
__device__ __forceinline__ v16bf a_frag(const __bf16* tile, int tileRow, int lane) {
  return load_frag_rows(tile + (size_t)(tileRow + (lane & 15)) * 32, (lane & 16) ? 8 : 0);
}
// B tile transposed in LDS: [64 n][BT_PITCH k]; lane col = cOff + (lane&15)
__device__ __forceinline__ v16bf b_frag(const __bf16* tileT, int cOff, int lane) {
  return load_frag_rows(tileT + (size_t)(cOff + (lane & 15)) * BT_PITCH, (lane & 16) ? 8 : 0);
}

#define WMMA_BF16(A, B, C) \
  __builtin_amdgcn_wmma_f32_16x16x32_bf16(false, (A), false, (B), (short)0, (C), false, false)

// ---------------- small prep kernels ----------------
__global__ void k_cvt_bf16(const float* __restrict__ src, __bf16* __restrict__ dst, int n) {
  int i = blockIdx.x * blockDim.x + threadIdx.x;
  if (i < n) dst[i] = f2bf(src[i]);
}

// W_out [1566,768] f32 -> transposed bf16 [768 n][1566 k]
__global__ void k_cvt_woutT(const float* __restrict__ src, __bf16* __restrict__ dst) {
  int idx = blockIdx.x * blockDim.x + threadIdx.x;
  if (idx >= 768 * 1566) return;
  int n = idx / 1566, k = idx % 1566;
  dst[idx] = f2bf(src[(size_t)k * H_DIM + n]);
}

// W_p1 [3072,150] -> transposed, padded: W1acT/W1bcT/W1dT each [PHPAD n][768 k] bf16
__global__ void k_make_w1T(const float* __restrict__ Wp1,
                           __bf16* __restrict__ ac, __bf16* __restrict__ bc, __bf16* __restrict__ dd) {
  int idx = blockIdx.x * blockDim.x + threadIdx.x;
  if (idx >= PHPAD * H_DIM) return;
  int c = idx / H_DIM, k = idx % H_DIM;
  float a = 0.f, b = 0.f, d = 0.f;
  if (c < PHID) {
    float wa = Wp1[(size_t)k * PHID + c];
    float wb = Wp1[(size_t)(768 + k) * PHID + c];
    float wc = Wp1[(size_t)(1536 + k) * PHID + c];
    float wd = Wp1[(size_t)(2304 + k) * PHID + c];
    a = wa + wc; b = wb - wc; d = wd;
  }
  ac[idx] = f2bf(a); bc[idx] = f2bf(b); dd[idx] = f2bf(d);
}

// width_table[w][n] = width_emb[w]@W_out_tail + b_out + b_start@W_out_top + b_end@W_out_mid
__global__ void k_width_table(const float* __restrict__ W_out, const float* __restrict__ b_start,
                              const float* __restrict__ b_end, const float* __restrict__ b_out,
                              const float* __restrict__ wemb, float* __restrict__ wtab) {
  int n = blockIdx.x * blockDim.x + threadIdx.x;
  if (n >= H_DIM) return;
  float c = b_out[n];
  for (int k = 0; k < H_DIM; ++k) {
    c += b_start[k] * W_out[(size_t)k * H_DIM + n];
    c += b_end[k]   * W_out[(size_t)(H_DIM + k) * H_DIM + n];
  }
  for (int w = 0; w <= MAXW; ++w) {
    float s = c;
    for (int d = 0; d < 30; ++d) s += wemb[w * 30 + d] * W_out[(size_t)(1536 + d) * H_DIM + n];
    wtab[w * H_DIM + n] = s;
  }
}

__global__ void k_init_scores(float* __restrict__ s, const float* __restrict__ bm, int n) {
  int i = blockIdx.x * blockDim.x + threadIdx.x;
  if (i < n) s[i] = bm[0];
}

// ---------------- generic bf16 WMMA GEMM: C[M,N] = A[M,K] @ BT^T ----------------
// A row-major [M,K] bf16; BT is B transposed: [N][ldB] bf16 (row n = output col).
// OUT_T == 0: write f32 row-major [M,N].  OUT_T == 1: write bf16 transposed [N][M].
template <int OUT_T>
__global__ __launch_bounds__(256) void k_gemm_bf16(
    const __bf16* __restrict__ A, const __bf16* __restrict__ BT, int ldB,
    int M, int K, int N, float* __restrict__ outF, __bf16* __restrict__ outBfT) {
  __shared__ __bf16 sA[128 * 32];
  __shared__ __bf16 sBT[64 * BT_PITCH];
  int tid = threadIdx.x, lane = tid & 31, wave = tid >> 5;
  int wm = wave >> 1, wn = wave & 1;
  int m0 = blockIdx.x * 128, n0 = blockIdx.y * 64;
  v8f acc[2][2] = {};
  int r = tid >> 1, kb = (tid & 1) * 16;      // A staging: 2 threads per row
  int nn = tid >> 2, kc = (tid & 3) * 8;      // B staging: 4 threads per BT row (16B each)
  for (int k0 = 0; k0 < K; k0 += 32) {
    int row = m0 + r; if (row >= M) row = M - 1;
    const __bf16* pa = A + (size_t)row * K + k0 + kb;
    *(v8bf*)&sA[r * 32 + kb]     = *(const v8bf*)(pa);
    *(v8bf*)&sA[r * 32 + kb + 8] = *(const v8bf*)(pa + 8);
    async_copy16(BT + (size_t)(n0 + nn) * ldB + k0 + kc, &sBT[nn * BT_PITCH + kc]);
    async_wait0();
    __syncthreads();
    v16bf af[2], bf[2];
#pragma unroll
    for (int t = 0; t < 2; ++t) {
      af[t] = a_frag(sA, wm * 32 + t * 16, lane);
      bf[t] = b_frag(sBT, wn * 32 + t * 16, lane);
    }
#pragma unroll
    for (int tm = 0; tm < 2; ++tm)
#pragma unroll
      for (int tn = 0; tn < 2; ++tn)
        acc[tm][tn] = WMMA_BF16(af[tm], bf[tn], acc[tm][tn]);
    __syncthreads();
  }
  int colw = lane & 15, rowh = (lane & 16) ? 8 : 0;
#pragma unroll
  for (int tm = 0; tm < 2; ++tm)
#pragma unroll
    for (int tn = 0; tn < 2; ++tn)
#pragma unroll
      for (int e = 0; e < 8; ++e) {
        int row = m0 + wm * 32 + tm * 16 + rowh + e;
        int col = n0 + wn * 32 + tn * 16 + colw;
        if (row < M) {
          float v = acc[tm][tn][e];
          if (OUT_T == 0) outF[(size_t)row * N + col] = v;
          else            outBfT[(size_t)col * M + row] = f2bf(v);
        }
      }
}

// ---------------- fused span GEMM (double-buffered, async-pipelined) ----------------
// C = gather(seq,start)@WcS + gather(seq,end)@WcE (+ width_table row); Wc given transposed.
// MODE 0: epilogue reduces against w_mention -> atomicAdd into scores[row] (C never stored)
// MODE 1: rows indirected via row_map (top-k), writes bf16 repr_k row-major
template <int MODE>
__global__ __launch_bounds__(256) void k_span_gemm(
    const float* __restrict__ seq, const int* __restrict__ starts, const int* __restrict__ ends,
    const int* __restrict__ row_map, int M,
    const __bf16* __restrict__ WcST, const __bf16* __restrict__ WcET,
    const float* __restrict__ wtab, const float* __restrict__ wmen,
    float* __restrict__ scores, __bf16* __restrict__ reprOut) {
  __shared__ __bf16 sAs[2][128 * 32];
  __shared__ __bf16 sAe[2][128 * 32];
  __shared__ __bf16 sBTs[2][64 * BT_PITCH];
  __shared__ __bf16 sBTe[2][64 * BT_PITCH];
  __shared__ long long offS[128], offE[128];
  __shared__ int wdt[128];
  __shared__ float rowsum[128];
  int tid = threadIdx.x, lane = tid & 31, wave = tid >> 5;
  int wm = wave >> 1, wn = wave & 1;
  int m0 = blockIdx.x * 128, n0 = blockIdx.y * 64;
  if (tid < 128) {
    int row = m0 + tid; if (row >= M) row = M - 1;
    int b, span;
    if (MODE == 1) { b = row / TOPK; span = b * NSPAN + row_map[row]; }
    else           { span = row; b = row >> 13; }
    int st = starts[span], en = ends[span];
    int w = en - st; w = w < 0 ? 0 : (w > MAXW ? MAXW : w);
    offS[tid] = ((long long)b * SEQ_S + st) * H_DIM;
    offE[tid] = ((long long)b * SEQ_S + en) * H_DIM;
    wdt[tid] = w;
  }
  __syncthreads();
  v8f acc[2][2] = {};
  int r = tid >> 1, kb = (tid & 1) * 16;
  int nn = tid >> 2, kc = (tid & 3) * 8;

  float4 aS[4], aE[4];                       // register pipeline for gathered A rows
  auto loadA = [&](int k0) {
    const float* ps = seq + offS[r] + k0 + kb;
    const float* pe = seq + offE[r] + k0 + kb;
#pragma unroll
    for (int q = 0; q < 4; ++q) {
      aS[q] = *(const float4*)(ps + 4 * q);
      aE[q] = *(const float4*)(pe + 4 * q);
    }
    if (k0 + 64 < H_DIM) {                   // prefetch tile after next (global_prefetch_b8)
      __builtin_prefetch(ps + 64, 0, 1);
      __builtin_prefetch(pe + 64, 0, 1);
    }
  };
  auto storeA = [&](int buf) {
    __bf16* dS = &sAs[buf][r * 32 + kb];
    __bf16* dE = &sAe[buf][r * 32 + kb];
    *(v8bf*)dS       = pack8(aS[0], aS[1]);
    *(v8bf*)(dS + 8) = pack8(aS[2], aS[3]);
    *(v8bf*)dE       = pack8(aE[0], aE[1]);
    *(v8bf*)(dE + 8) = pack8(aE[2], aE[3]);
  };
  auto issueB = [&](int k0, int buf) {
    async_copy16(WcST + (size_t)(n0 + nn) * H_DIM + k0 + kc, &sBTs[buf][nn * BT_PITCH + kc]);
    async_copy16(WcET + (size_t)(n0 + nn) * H_DIM + k0 + kc, &sBTe[buf][nn * BT_PITCH + kc]);
  };

  // prologue: stage tile 0, start loading tile 1 into registers
  loadA(0);
  issueB(0, 0);
  storeA(0);
  loadA(32);

  for (int it = 0; it < KTILES; ++it) {
    int p = it & 1, q = p ^ 1;
    async_wait0();                 // tile 'it' B copies complete (issued one stage ago)
    __syncthreads();               // tile 'it' A stores visible; buffer q free for reuse
    if (it + 1 < KTILES) issueB((it + 1) * 32, q);   // overlaps with WMMA below
    v16bf as[2], ae[2], bs[2], be[2];
#pragma unroll
    for (int t = 0; t < 2; ++t) {
      as[t] = a_frag(sAs[p], wm * 32 + t * 16, lane);
      ae[t] = a_frag(sAe[p], wm * 32 + t * 16, lane);
      bs[t] = b_frag(sBTs[p], wn * 32 + t * 16, lane);
      be[t] = b_frag(sBTe[p], wn * 32 + t * 16, lane);
    }
#pragma unroll
    for (int tm = 0; tm < 2; ++tm)
#pragma unroll
      for (int tn = 0; tn < 2; ++tn) {
        acc[tm][tn] = WMMA_BF16(as[tm], bs[tn], acc[tm][tn]);
        acc[tm][tn] = WMMA_BF16(ae[tm], be[tn], acc[tm][tn]);
      }
    if (it + 1 < KTILES) {
      storeA(q);                   // registers hold tile it+1
      if (it + 2 < KTILES) loadA((it + 2) * 32);
    }
  }

  int colw = lane & 15, rowh = (lane & 16) ? 8 : 0;
  if (MODE == 1) {
#pragma unroll
    for (int tm = 0; tm < 2; ++tm)
#pragma unroll
      for (int tn = 0; tn < 2; ++tn)
#pragma unroll
        for (int e = 0; e < 8; ++e) {
          int rloc = wm * 32 + tm * 16 + rowh + e;
          int row = m0 + rloc;
          int col = n0 + wn * 32 + tn * 16 + colw;
          if (row < M) {
            float v = acc[tm][tn][e] + wtab[wdt[rloc] * H_DIM + col];
            reprOut[(size_t)row * H_DIM + col] = f2bf(v);
          }
        }
  } else {
    __syncthreads();
    if (tid < 128) {
      float s = 0.f; int w = wdt[tid];
      for (int j = 0; j < 64; ++j) s += wtab[w * H_DIM + n0 + j] * wmen[n0 + j];
      rowsum[tid] = s;
    }
    __syncthreads();
#pragma unroll
    for (int tm = 0; tm < 2; ++tm)
#pragma unroll
      for (int tn = 0; tn < 2; ++tn)
#pragma unroll
        for (int e = 0; e < 8; ++e) {
          int rloc = wm * 32 + tm * 16 + rowh + e;
          int col = n0 + wn * 32 + tn * 16 + colw;
          atomicAdd(&rowsum[rloc], acc[tm][tn][e] * wmen[col]);   // ds_add_f32
        }
    __syncthreads();
    if (tid < 128) {
      int row = m0 + tid;
      if (row < M) atomicAdd(&scores[row], rowsum[tid]);          // global_atomic_add_f32
    }
  }
}

// ---------------- pair product GEMM: Cp[p,:] = (s_i ⊙ s_j) @ W1d (W1d transposed) --------
__global__ __launch_bounds__(256) void k_pair_gemm(
    const __bf16* __restrict__ reprk, const __bf16* __restrict__ W1dT,
    float* __restrict__ Cp, int Mreal) {
  __shared__ __bf16 sA[128 * 32];
  __shared__ __bf16 sBT[64 * BT_PITCH];
  __shared__ int offI[128], offJ[128];
  int tid = threadIdx.x, lane = tid & 31, wave = tid >> 5;
  int wm = wave >> 1, wn = wave & 1;
  int m0 = blockIdx.x * 128, n0 = blockIdx.y * 64;
  if (tid < 128) {
    int p = m0 + tid; if (p >= Mreal) p = Mreal - 1;
    int b = p / (TOPK * TOPK), rr = p % (TOPK * TOPK);
    int i = rr / TOPK, j = rr % TOPK;
    offI[tid] = (b * TOPK + i) * H_DIM;
    offJ[tid] = (b * TOPK + j) * H_DIM;
  }
  __syncthreads();
  v8f acc[2][2] = {};
  int r = tid >> 1, kb = (tid & 1) * 16;
  int nn = tid >> 2, kc = (tid & 3) * 8;
  for (int k0 = 0; k0 < H_DIM; k0 += 32) {
    const __bf16* pi = reprk + offI[r] + k0 + kb;
    const __bf16* pj = reprk + offJ[r] + k0 + kb;
    v8bf c0v, c1v;
#pragma unroll
    for (int q = 0; q < 8; ++q) c0v[q] = f2bf(bf2f(pi[q]) * bf2f(pj[q]));
#pragma unroll
    for (int q = 0; q < 8; ++q) c1v[q] = f2bf(bf2f(pi[8 + q]) * bf2f(pj[8 + q]));
    *(v8bf*)&sA[r * 32 + kb] = c0v; *(v8bf*)&sA[r * 32 + kb + 8] = c1v;
    async_copy16(W1dT + (size_t)(n0 + nn) * H_DIM + k0 + kc, &sBT[nn * BT_PITCH + kc]);
    async_wait0();
    __syncthreads();
    v16bf af[2], bf[2];
#pragma unroll
    for (int t = 0; t < 2; ++t) {
      af[t] = a_frag(sA, wm * 32 + t * 16, lane);
      bf[t] = b_frag(sBT, wn * 32 + t * 16, lane);
    }
#pragma unroll
    for (int tm = 0; tm < 2; ++tm)
#pragma unroll
      for (int tn = 0; tn < 2; ++tn)
        acc[tm][tn] = WMMA_BF16(af[tm], bf[tn], acc[tm][tn]);
    __syncthreads();
  }
  int colw = lane & 15, rowh = (lane & 16) ? 8 : 0;
#pragma unroll
  for (int tm = 0; tm < 2; ++tm)
#pragma unroll
    for (int tn = 0; tn < 2; ++tn)
#pragma unroll
      for (int e = 0; e < 8; ++e) {
        int row = m0 + wm * 32 + tm * 16 + rowh + e;
        int col = n0 + wn * 32 + tn * 16 + colw;
        if (row < Mreal) Cp[(size_t)row * PHPAD + col] = acc[tm][tn][e];
      }
}

// ---------------- top-k per batch (iterative argmax in LDS) ----------------
__global__ __launch_bounds__(1024) void k_topk(const float* __restrict__ scores,
                                               float* __restrict__ topv, int* __restrict__ topi) {
  __shared__ float vals[NSPAN];
  __shared__ float rv[1024];
  __shared__ int   ri[1024];
  int b = blockIdx.x, t = threadIdx.x;
  for (int idx = t; idx < NSPAN; idx += 1024) vals[idx] = scores[b * NSPAN + idx];
  __syncthreads();
  for (int k = 0; k < TOPK; ++k) {
    float bv = -INFINITY; int bi = 0;
    for (int idx = t; idx < NSPAN; idx += 1024) {
      float v = vals[idx];
      if (v > bv) { bv = v; bi = idx; }
    }
    rv[t] = bv; ri[t] = bi; __syncthreads();
    for (int s = 512; s > 0; s >>= 1) {
      if (t < s) {
        if (rv[t + s] > rv[t] || (rv[t + s] == rv[t] && ri[t + s] < ri[t])) {
          rv[t] = rv[t + s]; ri[t] = ri[t + s];
        }
      }
      __syncthreads();
    }
    if (t == 0) {
      topv[b * TOPK + k] = rv[0];
      topi[b * TOPK + k] = ri[0];
      vals[ri[0]] = -INFINITY;
    }
    __syncthreads();
  }
}

// ---------------- finalize: relu + layernorm + W_p2 dot + mask ----------------
__global__ __launch_bounds__(256) void k_finalize(
    const float* __restrict__ Cp, const float* __restrict__ U, const float* __restrict__ V,
    const float* __restrict__ b_p1, const float* __restrict__ g, const float* __restrict__ bb,
    const float* __restrict__ W_p2, const float* __restrict__ b_p2,
    float* __restrict__ ant, float* __restrict__ mask) {
  __shared__ float red[256];
  int p = blockIdx.x;
  int b = p / (TOPK * TOPK), rr = p % (TOPK * TOPK), i = rr / TOPK, j = rr % TOPK;
  int t = threadIdx.x;
  float v = 0.f;
  if (t < PHID) {
    v = Cp[(size_t)p * PHPAD + t] + U[(size_t)(b * TOPK + i) * PHPAD + t]
      + V[(size_t)(b * TOPK + j) * PHPAD + t] + b_p1[t];
    v = v > 0.f ? v : 0.f;
  }
  red[t] = v; __syncthreads();
  for (int s = 128; s > 0; s >>= 1) { if (t < s) red[t] += red[t + s]; __syncthreads(); }
  float mu = red[0] / PHID; __syncthreads();
  float d = (t < PHID) ? (v - mu) : 0.f;
  red[t] = d * d; __syncthreads();
  for (int s = 128; s > 0; s >>= 1) { if (t < s) red[t] += red[t + s]; __syncthreads(); }
  float var = red[0] / PHID; __syncthreads();
  float rs = rsqrtf(var + 1e-5f);
  float contrib = 0.f;
  if (t < PHID) contrib = (d * rs * g[t] + bb[t]) * W_p2[t];
  red[t] = contrib; __syncthreads();
  for (int s = 128; s > 0; s >>= 1) { if (t < s) red[t] += red[t + s]; __syncthreads(); }
  if (t == 0) {
    float score = red[0] + b_p2[0];
    ant[p] = (j < i) ? score : -INFINITY;
    if (b == 0) mask[rr] = (j < i) ? 1.f : 0.f;
  }
}

// ---------------- host launch ----------------
extern "C" void kernel_launch(void* const* d_in, const int* in_sizes, int n_in,
                              void* d_out, int out_size, void* d_ws, size_t ws_size,
                              hipStream_t stream) {
  const float* seq       = (const float*)d_in[0];
  const int*   starts    = (const int*)d_in[1];
  const int*   ends      = (const int*)d_in[2];
  const float* W_start   = (const float*)d_in[3];
  const float* b_start   = (const float*)d_in[4];
  const float* W_end     = (const float*)d_in[5];
  const float* b_end     = (const float*)d_in[6];
  const float* width_emb = (const float*)d_in[7];
  const float* W_out     = (const float*)d_in[8];
  const float* b_out     = (const float*)d_in[9];
  const float* w_mention = (const float*)d_in[10];
  const float* b_mention = (const float*)d_in[11];
  const float* W_p1      = (const float*)d_in[12];
  const float* b_p1      = (const float*)d_in[13];
  const float* ln_g      = (const float*)d_in[14];
  const float* ln_b      = (const float*)d_in[15];
  const float* W_p2      = (const float*)d_in[16];
  const float* b_p2      = (const float*)d_in[17];
  // d_in[18]: top_k scalar (compile-time TOPK)

  char* ws = (char*)d_ws;
  size_t off = 0;
  auto alloc = [&](size_t bytes) -> void* {
    void* p = (void*)(ws + off);
    off += (bytes + 255) & ~(size_t)255;
    return p;
  };
  __bf16* WstartBf = (__bf16*)alloc((size_t)768 * 768 * 2);
  __bf16* WendBf   = (__bf16*)alloc((size_t)768 * 768 * 2);
  __bf16* WoutT    = (__bf16*)alloc((size_t)768 * 1566 * 2);   // [n][k] transposed
  __bf16* WcST     = (__bf16*)alloc((size_t)768 * 768 * 2);    // [n][k]
  __bf16* WcET     = (__bf16*)alloc((size_t)768 * 768 * 2);    // [n][k]
  float*  wtab     = (float*) alloc((size_t)11 * 768 * 4);
  __bf16* W1acT    = (__bf16*)alloc((size_t)PHPAD * 768 * 2);  // [n][k]
  __bf16* W1bcT    = (__bf16*)alloc((size_t)PHPAD * 768 * 2);
  __bf16* W1dT     = (__bf16*)alloc((size_t)PHPAD * 768 * 2);
  float*  scores   = (float*) alloc((size_t)NBATCH * NSPAN * 4);
  int*    topi     = (int*)   alloc((size_t)NBATCH * TOPK * 4);
  __bf16* reprk    = (__bf16*)alloc((size_t)NBATCH * TOPK * 768 * 2);
  float*  U        = (float*) alloc((size_t)NBATCH * TOPK * PHPAD * 4);
  float*  V        = (float*) alloc((size_t)NBATCH * TOPK * PHPAD * 4);
  float*  Cp       = (float*) alloc((size_t)NBATCH * TOPK * TOPK * PHPAD * 4);
  (void)ws_size; (void)n_in; (void)in_sizes; (void)out_size;

  float* out = (float*)d_out;          // [400 top_scores | 20000 ant_scores | 2500 mask]
  const int NWe = 768 * 768;

  // weight prep
  k_cvt_bf16<<<(NWe + 255) / 256, 256, 0, stream>>>(W_start, WstartBf, NWe);
  k_cvt_bf16<<<(NWe + 255) / 256, 256, 0, stream>>>(W_end, WendBf, NWe);
  k_cvt_woutT<<<(768 * 1566 + 255) / 256, 256, 0, stream>>>(W_out, WoutT);
  k_make_w1T<<<(PHPAD * 768 + 255) / 256, 256, 0, stream>>>(W_p1, W1acT, W1bcT, W1dT);
  k_width_table<<<3, 256, 0, stream>>>(W_out, b_start, b_end, b_out, width_emb, wtab);

  // fused weights (transposed outputs): WcS = W_start@W_out[0:768], WcE = W_end@W_out[768:1536]
  k_gemm_bf16<1><<<dim3(6, 12), 256, 0, stream>>>(WstartBf, WoutT,       1566, 768, 768, 768, nullptr, WcST);
  k_gemm_bf16<1><<<dim3(6, 12), 256, 0, stream>>>(WendBf,   WoutT + 768, 1566, 768, 768, 768, nullptr, WcET);

  // mention scores for all 65536 spans (span_repr never materialized)
  k_init_scores<<<256, 256, 0, stream>>>(scores, b_mention, NBATCH * NSPAN);
  k_span_gemm<0><<<dim3(512, 12), 256, 0, stream>>>(seq, starts, ends, nullptr, NBATCH * NSPAN,
                                                    WcST, WcET, wtab, w_mention, scores, nullptr);

  // top-k (writes top_scores directly to d_out)
  k_topk<<<NBATCH, 1024, 0, stream>>>(scores, out, topi);

  // recompute repr for the 400 selected spans
  k_span_gemm<1><<<dim3(4, 12), 256, 0, stream>>>(seq, starts, ends, topi, NBATCH * TOPK,
                                                  WcST, WcET, wtab, w_mention, scores, reprk);

  // rank-1 pair terms + elementwise-product pair GEMM
  k_gemm_bf16<0><<<dim3(4, 3), 256, 0, stream>>>(reprk, W1acT, 768, NBATCH * TOPK, 768, PHPAD, U, nullptr);
  k_gemm_bf16<0><<<dim3(4, 3), 256, 0, stream>>>(reprk, W1bcT, 768, NBATCH * TOPK, 768, PHPAD, V, nullptr);
  k_pair_gemm<<<dim3(157, 3), 256, 0, stream>>>(reprk, W1dT, Cp, NBATCH * TOPK * TOPK);

  // relu + layernorm + score + mask
  k_finalize<<<NBATCH * TOPK * TOPK, 256, 0, stream>>>(Cp, U, V, b_p1, ln_g, ln_b, W_p2, b_p2,
                                                       out + NBATCH * TOPK,
                                                       out + NBATCH * TOPK + NBATCH * TOPK * TOPK);
}
